// GCN_graph2_6090263626388
// MI455X (gfx1250) — matrix operand order repaired and checked by reading, not compiled, not run on hardware
//
#include <hip/hip_runtime.h>
#include <hip/hip_bf16.h>

typedef __attribute__((ext_vector_type(16))) _Float16 v16h;
typedef __attribute__((ext_vector_type(8)))  float    v8f;

#define HID 64
#define LN_EPS 1e-5f

// ---------------------------------------------------------------------------
// Zero-fill (workspace is poisoned 0xAA by the harness; atomics need zeros)
// ---------------------------------------------------------------------------
__global__ void zero_kernel(float* __restrict__ p, long long n) {
  long long i = (long long)blockIdx.x * blockDim.x + threadIdx.x;
  long long stride = (long long)gridDim.x * blockDim.x;
  for (; i < n; i += stride) p[i] = 0.0f;
}

// ---------------------------------------------------------------------------
// Degree via f32 atomics into dinv buffer (then converted in place)
// ---------------------------------------------------------------------------
__global__ void edge_degree_kernel(const int* __restrict__ col,
                                   float* __restrict__ deg, int E) {
  int e = blockIdx.x * blockDim.x + threadIdx.x;
  if (e < E) atomicAdd(&deg[col[e]], 1.0f);
}

__global__ void node_dinv_kernel(float* __restrict__ dinv, int N) {
  int n = blockIdx.x * blockDim.x + threadIdx.x;
  if (n < N) dinv[n] = rsqrtf(dinv[n] + 1.0f);   // +1 self-loop
}

// norm[e] = dinv[row]*dinv[col] — computed once, reused by all 3 layers
__global__ void edge_norm_kernel(const int* __restrict__ row,
                                 const int* __restrict__ col,
                                 const float* __restrict__ dinv,
                                 float* __restrict__ nrm, int E) {
  int e = blockIdx.x * blockDim.x + threadIdx.x;
  if (e < E) nrm[e] = dinv[row[e]] * dinv[col[e]];
}

// ---------------------------------------------------------------------------
// H[N x 64] = X[N x K] @ W[K x 64] via v_wmma_f32_16x16x32_f16 (wave32).
// Block = 128 threads = 4 waves. Block owns 16 rows; wave w owns cols 16w..16w+15.
// A 16x32 f16 layout (contiguous k): lane<16  -> a[0..15]=K(kb..kb+15)
//                                  lane>=16 -> a[0..15]=K(kb+16..kb+31)
// B 32x16 f16 layout: col = lane%16; lanes<16 hold K kb..kb+15 in b[0..15],
//                     lanes>=16 hold K kb+16..kb+31.
// D f32: lane holds col=lane%16; acc[j] is row j + 8*(lane/16).
// ---------------------------------------------------------------------------
__global__ __launch_bounds__(128) void gemm_wmma_kernel(
    const float* __restrict__ X, const float* __restrict__ W,
    float* __restrict__ H, int N, int K) {
  const int lane = threadIdx.x & 31;
  const int wave = threadIdx.x >> 5;        // 0..3 : column tile
  const int row0 = blockIdx.x * 16;
  if (row0 >= N) return;                    // uniform per block (EXEC stays full)
  const int half = lane >> 4;               // 0 | 1
  const int l16  = lane & 15;
  const int colN = wave * 16 + l16;         // output column 0..63

  v8f acc = {};
  for (int kb = 0; kb < K; kb += 32) {
    v16h a, b;
    const float* xrow = X + (size_t)(row0 + l16) * K + kb + half * 16;
#pragma unroll
    for (int h = 0; h < 16; ++h) {
      a[h]     = (_Float16)xrow[h];         // K = kb + 16*half + h
    }
    const float* wcol = W + (size_t)(kb + 16 * half) * HID + colN;
#pragma unroll
    for (int h = 0; h < 16; ++h) b[h] = (_Float16)wcol[h * HID];

    acc = __builtin_amdgcn_wmma_f32_16x16x32_f16(
        /*neg_a=*/false, a, /*neg_b=*/false, b,
        /*c_mod=*/(short)0, acc, /*reuse_a=*/false, /*reuse_b=*/false);
  }

  float* hout = H + (size_t)row0 * HID + colN;
#pragma unroll
  for (int j = 0; j < 8; ++j)
    hout[(size_t)(j + 8 * half) * HID] = acc[j];
}

// ---------------------------------------------------------------------------
// agg[col[e]] += norm[e] * h[row[e]]  — 16 threads/edge, float4 per thread.
// 128-bit coalesced gather + 4x global_atomic_add_f32 scatter (resolved in L2).
// ---------------------------------------------------------------------------
__global__ __launch_bounds__(256) void edge_agg_kernel(
    const int* __restrict__ row, const int* __restrict__ col,
    const float* __restrict__ nrm, const float* __restrict__ h,
    float* __restrict__ agg, int E) {
  long long gid = (long long)blockIdx.x * blockDim.x + threadIdx.x;
  int e = (int)(gid >> 4);
  if (e >= E) return;
  int c4 = ((int)gid & 15) * 4;
  float nm = nrm[e];
  const float4 v = *(const float4*)(h + (size_t)row[e] * HID + c4);
  float* dst = agg + (size_t)col[e] * HID + c4;
  atomicAdd(dst + 0, v.x * nm);
  atomicAdd(dst + 1, v.y * nm);
  atomicAdd(dst + 2, v.z * nm);
  atomicAdd(dst + 3, v.w * nm);
}

// ---------------------------------------------------------------------------
// Fused: agg += dinv^2 * h + bias ; LayerNorm over 64 feats ; ReLU. In place.
// One wave32 per node, 2 features per lane, shfl_xor reductions (no LDS).
// ---------------------------------------------------------------------------
__global__ __launch_bounds__(256) void node_ln_relu_kernel(
    const float* __restrict__ h, const float* __restrict__ dinv,
    const float* __restrict__ bias, const float* __restrict__ lnw,
    const float* __restrict__ lnb, float* __restrict__ agg, int N) {
  const int lane = threadIdx.x & 31;
  const int wave = threadIdx.x >> 5;
  const int n = blockIdx.x * 8 + wave;
  if (n >= N) return;
  const float di = dinv[n];
  const float self = di * di;
  const size_t base = (size_t)n * HID;

  float v0 = agg[base + lane]      + self * h[base + lane]      + bias[lane];
  float v1 = agg[base + lane + 32] + self * h[base + lane + 32] + bias[lane + 32];

  float s  = v0 + v1;
  float ss = v0 * v0 + v1 * v1;
#pragma unroll
  for (int o = 16; o >= 1; o >>= 1) {
    s  += __shfl_xor(s,  o, 32);
    ss += __shfl_xor(ss, o, 32);
  }
  const float mu  = s * (1.0f / 64.0f);
  const float var = ss * (1.0f / 64.0f) - mu * mu;
  const float inv = rsqrtf(var + LN_EPS);

  float y0 = (v0 - mu) * inv * lnw[lane]      + lnb[lane];
  float y1 = (v1 - mu) * inv * lnw[lane + 32] + lnb[lane + 32];
  agg[base + lane]      = y0 > 0.0f ? y0 : 0.0f;
  agg[base + lane + 32] = y1 > 0.0f ? y1 : 0.0f;
}

// ---------------------------------------------------------------------------
// pooled[f] = sum_n h[n][f]   (per-thread partials, one atomic per thread)
// ---------------------------------------------------------------------------
__global__ __launch_bounds__(256) void pool_kernel(
    const float* __restrict__ h, float* __restrict__ pooled, int N) {
  const int f = threadIdx.x & 63;
  const int grp = blockIdx.x * (blockDim.x >> 6) + (threadIdx.x >> 6);
  const int stride = gridDim.x * (blockDim.x >> 6);
  float s = 0.0f;
  for (int n = grp; n < N; n += stride) s += h[(size_t)n * HID + f];
  atomicAdd(&pooled[f], s);
}

// out[o] = (pooled/N) @ Wl[:,o] + bl[o]   — Wl is [64,25] row-major
__global__ void final_kernel(const float* __restrict__ pooled,
                             const float* __restrict__ Wl,
                             const float* __restrict__ bl,
                             float* __restrict__ out, float invN) {
  int o = threadIdx.x;
  if (o < 25) {
    float s = bl[o];
#pragma unroll
    for (int f = 0; f < HID; ++f) s += pooled[f] * invN * Wl[f * 25 + o];
    out[o] = s;
  }
}

// ---------------------------------------------------------------------------
extern "C" void kernel_launch(void* const* d_in, const int* in_sizes, int n_in,
                              void* d_out, int out_size, void* d_ws, size_t ws_size,
                              hipStream_t stream) {
  const float* x   = (const float*)d_in[0];
  const int*   ei  = (const int*)d_in[1];
  const float* W1  = (const float*)d_in[2];
  const float* b1  = (const float*)d_in[3];
  const float* W2  = (const float*)d_in[4];
  const float* b2  = (const float*)d_in[5];
  const float* W3  = (const float*)d_in[6];
  const float* b3  = (const float*)d_in[7];
  const float* ln1w = (const float*)d_in[8],  *ln1b = (const float*)d_in[9];
  const float* ln2w = (const float*)d_in[10], *ln2b = (const float*)d_in[11];
  const float* ln3w = (const float*)d_in[12], *ln3b = (const float*)d_in[13];
  const float* Wl  = (const float*)d_in[14];
  const float* bl  = (const float*)d_in[15];
  float* out = (float*)d_out;

  const int E = in_sizes[1] / 2;
  const int K1 = in_sizes[2] / HID;        // 128
  const int N = in_sizes[0] / K1;          // 100000
  const int* row = ei;
  const int* col = ei + E;

  // Workspace carve-up
  float* hbuf   = (float*)d_ws;            // [N*64] GEMM output
  float* abuf   = hbuf + (size_t)N * HID;  // [N*64] agg / features
  float* bbuf   = abuf + (size_t)N * HID;  // [N*64] agg / features
  float* dinv   = bbuf + (size_t)N * HID;  // [N]
  float* nrm    = dinv + N;                // [E]
  float* pooled = nrm + E;                 // [64]

  const int TPB = 256;
  const int edgeBlocks = (E + TPB - 1) / TPB;
  const int aggBlocks  = (int)(((long long)E * 16 + TPB - 1) / TPB);
  const int gemmBlocks = (N + 15) / 16;
  const int nodeBlocks = (N + 7) / 8;

  // ---- shared normalization (once, reused across layers) ----
  zero_kernel<<<1024, TPB, 0, stream>>>(dinv, (long long)N);
  edge_degree_kernel<<<edgeBlocks, TPB, 0, stream>>>(col, dinv, E);
  node_dinv_kernel<<<(N + TPB - 1) / TPB, TPB, 0, stream>>>(dinv, N);
  edge_norm_kernel<<<edgeBlocks, TPB, 0, stream>>>(row, col, dinv, nrm, E);

  // ---- layer 1: x[N,128] -> abuf ----
  gemm_wmma_kernel<<<gemmBlocks, 128, 0, stream>>>(x, W1, hbuf, N, K1);
  zero_kernel<<<4096, TPB, 0, stream>>>(abuf, (long long)N * HID);
  edge_agg_kernel<<<aggBlocks, TPB, 0, stream>>>(row, col, nrm, hbuf, abuf, E);
  node_ln_relu_kernel<<<nodeBlocks, TPB, 0, stream>>>(hbuf, dinv, b1, ln1w, ln1b, abuf, N);

  // ---- layer 2: abuf[N,64] -> bbuf ----
  gemm_wmma_kernel<<<gemmBlocks, 128, 0, stream>>>(abuf, W2, hbuf, N, HID);
  zero_kernel<<<4096, TPB, 0, stream>>>(bbuf, (long long)N * HID);
  edge_agg_kernel<<<aggBlocks, TPB, 0, stream>>>(row, col, nrm, hbuf, bbuf, E);
  node_ln_relu_kernel<<<nodeBlocks, TPB, 0, stream>>>(hbuf, dinv, b2, ln2w, ln2b, bbuf, N);

  // ---- layer 3: bbuf[N,64] -> abuf ----
  gemm_wmma_kernel<<<gemmBlocks, 128, 0, stream>>>(bbuf, W3, hbuf, N, HID);
  zero_kernel<<<4096, TPB, 0, stream>>>(abuf, (long long)N * HID);
  edge_agg_kernel<<<aggBlocks, TPB, 0, stream>>>(row, col, nrm, hbuf, abuf, E);
  node_ln_relu_kernel<<<nodeBlocks, TPB, 0, stream>>>(hbuf, dinv, b3, ln3w, ln3b, abuf, N);

  // ---- mean pool + final linear ----
  zero_kernel<<<1, 64, 0, stream>>>(pooled, 64);
  pool_kernel<<<512, TPB, 0, stream>>>(abuf, pooled, N);
  final_kernel<<<1, 32, 0, stream>>>(pooled, Wl, bl, out, 1.0f / (float)N);
}
